// SchNet_76587856823106
// MI455X (gfx1250) — matrix operand order repaired and checked by reading, not compiled, not run on hardware
//
#include <hip/hip_runtime.h>

#define Nn 50000
#define Ee 800000
#define Hh 128
#define Ff 128
#define Gg 50
#define Ll 6
#define H2v 64
#define OUTv 12
#define CUTOFF 10.0f
#define GSTEP (CUTOFF / (float)(Gg - 1))
#define GCOEFF (-0.5f / (GSTEP * GSTEP))
#define PIf 3.14159265358979f
#define LN2f 0.6931471805599453f

typedef __attribute__((ext_vector_type(16))) __bf16 v16bf;
typedef __attribute__((ext_vector_type(8)))  __bf16 v8bf;
typedef __attribute__((ext_vector_type(8)))  float  v8f;
typedef __attribute__((ext_vector_type(16))) float  v16f;
typedef __attribute__((ext_vector_type(4)))  float  v4f;

// ---------- helpers ----------

__device__ __forceinline__ float sspf(float x) {
    // softplus(x) - ln2, numerically stable
    float t = __logf(1.0f + __expf(-fabsf(x)));
    return fmaxf(x, 0.0f) + t - LN2f;
}

__device__ __forceinline__ v8f wmma_bf16(v16bf a, v16bf b, v8f c) {
    return __builtin_amdgcn_wmma_f32_16x16x32_bf16(
        /*neg_a=*/false, a, /*neg_b=*/false, b,
        /*c_mod=*/(short)0, c, /*reuse_a=*/false, /*reuse_b=*/false);
}

// B fragment: weights pre-swizzled so each lane reads 16 contiguous bf16 (32B).
// Block for (kt,nt) holds B[kt*32 + (e>>3)*16 + (lane>>4)*8 + (e&7)][nt*16 + (lane&15)].
__device__ __forceinline__ v16bf load_bfrag(const __bf16* w, int kt, int nt, int ntn, int lane) {
    const v16bf* p = (const v16bf*)(w + ((((size_t)kt * ntn + nt) << 5) + lane) * 16);
    return *p;
}

// A fragment from fp32 row-major [.,lda]: lane holds row m=lane&15,
// K = k0 + half*8 + {0..7} and K = k0 + 16 + half*8 + {0..7}.
__device__ __forceinline__ v16bf load_afrag_f32(const float* a, int row0, int lda, int k0, int lane) {
    int m = lane & 15, half = lane >> 4;
    const float* r = a + (size_t)(row0 + m) * lda + k0 + half * 8;
    v4f q0 = ((const v4f*)r)[0];
    v4f q1 = ((const v4f*)r)[1];
    v4f q2 = ((const v4f*)(r + 16))[0];
    v4f q3 = ((const v4f*)(r + 16))[1];
    v16f t;
    t[0]  = q0[0]; t[1]  = q0[1]; t[2]  = q0[2]; t[3]  = q0[3];
    t[4]  = q1[0]; t[5]  = q1[1]; t[6]  = q1[2]; t[7]  = q1[3];
    t[8]  = q2[0]; t[9]  = q2[1]; t[10] = q2[2]; t[11] = q2[3];
    t[12] = q3[0]; t[13] = q3[1]; t[14] = q3[2]; t[15] = q3[3];
    return __builtin_convertvector(t, v16bf);   // packed v_cvt_pk_bf16_f32
}

// A fragment from a per-wave bf16 LDS tile [16][ldt]
__device__ __forceinline__ v16bf load_afrag_lds(const __bf16* t, int ldt, int k0, int lane) {
    int m = lane & 15, half = lane >> 4;
    const __bf16* r = t + m * ldt + k0 + half * 8;
    v8bf lo = *(const v8bf*)r;
    v8bf hi = *(const v8bf*)(r + 16);
    v16bf a;
#pragma unroll
    for (int i = 0; i < 8; i++) { a[i] = lo[i]; a[8 + i] = hi[i]; }
    return a;
}

// Gaussian-smearing A fragment computed analytically from the edge distance.
__device__ __forceinline__ v16bf edge_afrag(float d, int kt, int half) {
    v16f t;
#pragma unroll
    for (int e = 0; e < 16; e++) {
        int g = kt * 32 + (e & 7) + ((e >> 3) << 4) + half * 8;
        float v = 0.0f;
        if (g < Gg) { float x = d - (float)g * GSTEP; v = __expf(GCOEFF * x * x); }
        t[e] = v;
    }
    return __builtin_convertvector(t, v16bf);
}

// ---------- kernels ----------

// Convert fp32 [K x (ntn*16)] row-major into bf16 B-fragment blocks (K padded to kt*32).
__global__ void k_swizzle(const float* __restrict__ src, __bf16* __restrict__ dst,
                          int K, int ntn, int total) {
    int o = blockIdx.x * blockDim.x + threadIdx.x;
    if (o >= total) return;
    int e = o & 15;
    int lane = (o >> 4) & 31;
    int blk = o >> 9;
    int nt = blk % ntn;
    int kt = blk / ntn;
    int krow = kt * 32 + (e & 7) + ((e >> 3) << 4) + (lane >> 4) * 8;
    int ncol = nt * 16 + (lane & 15);
    float v = (krow < K) ? src[(size_t)krow * (ntn * 16) + ncol] : 0.0f;
    dst[o] = (__bf16)v;
}

__global__ void k_embed(const int* __restrict__ xa, const float* __restrict__ emb,
                        float* __restrict__ h, int total) {
    int i = blockIdx.x * blockDim.x + threadIdx.x;
    if (i >= total) return;
    int node = i >> 7, c = i & 127;
    h[i] = emb[(size_t)xa[node] * Hh + c];
}

__global__ void k_zero(float* __restrict__ p, int n) {
    int i = blockIdx.x * blockDim.x + threadIdx.x;
    if (i < n) p[i] = 0.0f;
}

// xf = h @ conv1_w  (no bias), per-wave 16x128 tile, K=128
__global__ __launch_bounds__(128) void k_xf(const float* __restrict__ h,
                                            const __bf16* __restrict__ w,
                                            float* __restrict__ xf, int ntiles) {
    int lane = threadIdx.x & 31;
    int tile = blockIdx.x * 4 + (threadIdx.x >> 5);
    if (tile >= ntiles) return;
    int row0 = tile * 16, m = lane & 15, half = lane >> 4;
    v8f acc[8];
#pragma unroll
    for (int nt = 0; nt < 8; nt++)
#pragma unroll
        for (int r = 0; r < 8; r++) acc[nt][r] = 0.0f;
#pragma unroll
    for (int kt = 0; kt < 4; kt++) {
        v16bf a = load_afrag_f32(h, row0, 128, kt * 32, lane);
#pragma unroll
        for (int nt = 0; nt < 8; nt++)
            acc[nt] = wmma_bf16(a, load_bfrag(w, kt, nt, 8, lane), acc[nt]);
    }
#pragma unroll
    for (int nt = 0; nt < 8; nt++)
#pragma unroll
        for (int r = 0; r < 8; r++)
            xf[(size_t)(row0 + r + half * 8) * 128 + nt * 16 + m] = acc[nt][r];
}

// Per-edge filter MLP + gather + modulate + scatter-add.
// One wave = 16 edges x full 128 filter dims.
__global__ __launch_bounds__(128) void k_edge(const float* __restrict__ edge_attr,
                                              const int* __restrict__ eidx,
                                              const float* __restrict__ xf,
                                              float* __restrict__ agg,
                                              const __bf16* __restrict__ w1,
                                              const __bf16* __restrict__ w2,
                                              const float* __restrict__ b1,
                                              const float* __restrict__ b2,
                                              int ntiles) {
    __shared__ __bf16 ldsT[4][16 * 128];
    __shared__ float  ldsC[4][16];
    int lane = threadIdx.x & 31;
    int wib  = threadIdx.x >> 5;
    int tile = blockIdx.x * 4 + wib;
    if (tile >= ntiles) return;
    int e0 = tile * 16, m = lane & 15, half = lane >> 4;

    float d = edge_attr[e0 + m];
    if (half == 0) ldsC[wib][m] = 0.5f * (__cosf(d * (PIf / CUTOFF)) + 1.0f);

    v16bf a0 = edge_afrag(d, 0, half);
    v16bf a1 = edge_afrag(d, 1, half);

    // stage 1: ssp(ed @ mlp_w1 + b1) -> LDS bf16 tile
#pragma unroll
    for (int nt = 0; nt < 8; nt++) {
        float bias = b1[nt * 16 + m];
        v8f c;
#pragma unroll
        for (int r = 0; r < 8; r++) c[r] = bias;
        c = wmma_bf16(a0, load_bfrag(w1, 0, nt, 8, lane), c);
        c = wmma_bf16(a1, load_bfrag(w1, 1, nt, 8, lane), c);
#pragma unroll
        for (int r = 0; r < 8; r++)
            ldsT[wib][(r + half * 8) * 128 + nt * 16 + m] = (__bf16)sspf(c[r]);
    }
    asm volatile("s_wait_dscnt 0" ::: "memory");   // same-wave DS RAW fence

    // stage 2: W = T1 @ mlp_w2 + b2
    v8f acc[8];
#pragma unroll
    for (int nt = 0; nt < 8; nt++) {
        float bias = b2[nt * 16 + m];
#pragma unroll
        for (int r = 0; r < 8; r++) acc[nt][r] = bias;
    }
#pragma unroll
    for (int kt = 0; kt < 4; kt++) {
        v16bf a = load_afrag_lds(&ldsT[wib][0], 128, kt * 32, lane);
#pragma unroll
        for (int nt = 0; nt < 8; nt++)
            acc[nt] = wmma_bf16(a, load_bfrag(w2, kt, nt, 8, lane), acc[nt]);
    }

    // gather xf[src], modulate by W * cutoff, scatter-add to agg[dst]
    const int* srcp = eidx;
    const int* dstp = eidx + Ee;
#pragma unroll
    for (int r = 0; r < 8; r++) {
        int row = r + half * 8;
        int e = e0 + row;
        int s = srcp[e];
        int t = dstp[e];
        float cw = ldsC[wib][row];
        const float* xr = xf + (size_t)s * 128;
        float* ar = agg + (size_t)t * 128;
#pragma unroll
        for (int nt = 0; nt < 8; nt++) {
            int f = nt * 16 + m;
            float msg = acc[nt][r] * cw * xr[f];
            atomicAdd(&ar[f], msg);     // non-returning global_atomic_add_f32
        }
    }
}

// v = ssp(agg@conv2_w + b2) @ int_lin_w + b3 ; h += v
__global__ __launch_bounds__(128) void k_node_update(const float* __restrict__ agg,
                                                     const __bf16* __restrict__ w2,
                                                     const float* __restrict__ b2,
                                                     const __bf16* __restrict__ w3,
                                                     const float* __restrict__ b3,
                                                     float* __restrict__ h, int ntiles) {
    __shared__ __bf16 ldsT[4][16 * 128];
    int lane = threadIdx.x & 31;
    int wib  = threadIdx.x >> 5;
    int tile = blockIdx.x * 4 + wib;
    if (tile >= ntiles) return;
    int row0 = tile * 16, m = lane & 15, half = lane >> 4;

    v8f acc[8];
#pragma unroll
    for (int nt = 0; nt < 8; nt++) {
        float b = b2[nt * 16 + m];
#pragma unroll
        for (int r = 0; r < 8; r++) acc[nt][r] = b;
    }
#pragma unroll
    for (int kt = 0; kt < 4; kt++) {
        v16bf a = load_afrag_f32(agg, row0, 128, kt * 32, lane);
#pragma unroll
        for (int nt = 0; nt < 8; nt++)
            acc[nt] = wmma_bf16(a, load_bfrag(w2, kt, nt, 8, lane), acc[nt]);
    }
#pragma unroll
    for (int nt = 0; nt < 8; nt++)
#pragma unroll
        for (int r = 0; r < 8; r++)
            ldsT[wib][(r + half * 8) * 128 + nt * 16 + m] = (__bf16)sspf(acc[nt][r]);
    asm volatile("s_wait_dscnt 0" ::: "memory");

#pragma unroll
    for (int nt = 0; nt < 8; nt++) {
        float b = b3[nt * 16 + m];
#pragma unroll
        for (int r = 0; r < 8; r++) acc[nt][r] = b;
    }
#pragma unroll
    for (int kt = 0; kt < 4; kt++) {
        v16bf a = load_afrag_lds(&ldsT[wib][0], 128, kt * 32, lane);
#pragma unroll
        for (int nt = 0; nt < 8; nt++)
            acc[nt] = wmma_bf16(a, load_bfrag(w3, kt, nt, 8, lane), acc[nt]);
    }
#pragma unroll
    for (int nt = 0; nt < 8; nt++)
#pragma unroll
        for (int r = 0; r < 8; r++) {
            size_t idx = (size_t)(row0 + r + half * 8) * 128 + nt * 16 + m;
            h[idx] += acc[nt][r];
        }
}

// head: ssp(h@lin1+b1) @ lin2 + b2, column-sum into 64-wide accumulator
__global__ __launch_bounds__(128) void k_head(const float* __restrict__ h,
                                              const __bf16* __restrict__ w1,
                                              const float* __restrict__ b1,
                                              const __bf16* __restrict__ w2,
                                              const float* __restrict__ b2,
                                              float* __restrict__ hacc, int ntiles) {
    __shared__ __bf16 ldsT[4][16 * 64];
    int lane = threadIdx.x & 31;
    int wib  = threadIdx.x >> 5;
    int tile = blockIdx.x * 4 + wib;
    if (tile >= ntiles) return;
    int row0 = tile * 16, m = lane & 15, half = lane >> 4;

    v8f acc[4];
#pragma unroll
    for (int nt = 0; nt < 4; nt++) {
        float b = b1[nt * 16 + m];
#pragma unroll
        for (int r = 0; r < 8; r++) acc[nt][r] = b;
    }
#pragma unroll
    for (int kt = 0; kt < 4; kt++) {
        v16bf a = load_afrag_f32(h, row0, 128, kt * 32, lane);
#pragma unroll
        for (int nt = 0; nt < 4; nt++)
            acc[nt] = wmma_bf16(a, load_bfrag(w1, kt, nt, 4, lane), acc[nt]);
    }
#pragma unroll
    for (int nt = 0; nt < 4; nt++)
#pragma unroll
        for (int r = 0; r < 8; r++)
            ldsT[wib][(r + half * 8) * 64 + nt * 16 + m] = (__bf16)sspf(acc[nt][r]);
    asm volatile("s_wait_dscnt 0" ::: "memory");

#pragma unroll
    for (int nt = 0; nt < 4; nt++) {
        float b = b2[nt * 16 + m];
#pragma unroll
        for (int r = 0; r < 8; r++) acc[nt][r] = b;
    }
#pragma unroll
    for (int kt = 0; kt < 2; kt++) {
        v16bf a = load_afrag_lds(&ldsT[wib][0], 64, kt * 32, lane);
#pragma unroll
        for (int nt = 0; nt < 4; nt++)
            acc[nt] = wmma_bf16(a, load_bfrag(w2, kt, nt, 4, lane), acc[nt]);
    }
#pragma unroll
    for (int nt = 0; nt < 4; nt++) {
        float s = 0.0f;
#pragma unroll
        for (int r = 0; r < 8; r++) s += acc[nt][r];
        atomicAdd(&hacc[nt * 16 + m], s);   // both halves contribute their 8 rows
    }
}

__global__ void k_final(const float* __restrict__ acc, const float* __restrict__ rw,
                        const float* __restrict__ rb, float* __restrict__ out) {
    int j = threadIdx.x;
    if (j < OUTv) {
        float s = rb[j];
        for (int f = 0; f < H2v; f++) s += acc[f] * rw[f * OUTv + j];
        out[j] = s;
    }
}

// ---------- host ----------

extern "C" void kernel_launch(void* const* d_in, const int* in_sizes, int n_in,
                              void* d_out, int out_size, void* d_ws, size_t ws_size,
                              hipStream_t stream) {
    (void)in_sizes; (void)n_in; (void)out_size; (void)ws_size;

    const int*   x_atoms    = (const int*)  d_in[0];
    const int*   edge_index = (const int*)  d_in[1];
    const float* edge_attr  = (const float*)d_in[2];
    const float* embedding  = (const float*)d_in[3];
    const float* mlp_w1     = (const float*)d_in[4];
    const float* mlp_b1     = (const float*)d_in[5];
    const float* mlp_w2     = (const float*)d_in[6];
    const float* mlp_b2     = (const float*)d_in[7];
    const float* conv1_w    = (const float*)d_in[8];
    const float* conv2_w    = (const float*)d_in[9];
    const float* conv2_b    = (const float*)d_in[10];
    const float* int_lin_w  = (const float*)d_in[11];
    const float* int_lin_b  = (const float*)d_in[12];
    const float* lin1_w     = (const float*)d_in[13];
    const float* lin1_b     = (const float*)d_in[14];
    const float* lin2_w     = (const float*)d_in[15];
    const float* lin2_b     = (const float*)d_in[16];
    const float* readout_w  = (const float*)d_in[17];
    const float* readout_b  = (const float*)d_in[18];
    float* out = (float*)d_out;

    char* p = (char*)d_ws;
    auto alloc = [&](size_t b) -> void* {
        void* r = (void*)p;
        p += (b + 255) & ~(size_t)255;
        return r;
    };
    float*  h    = (float*)alloc((size_t)Nn * Hh * 4);
    float*  xf   = (float*)alloc((size_t)Nn * Ff * 4);
    float*  agg  = (float*)alloc((size_t)Nn * Ff * 4);
    float*  hacc = (float*)alloc(64 * 4);
    __bf16* w1s  = (__bf16*)alloc((size_t)Ll * 64 * 128 * 2);
    __bf16* w2s  = (__bf16*)alloc((size_t)Ll * 128 * 128 * 2);
    __bf16* c1s  = (__bf16*)alloc((size_t)Ll * 128 * 128 * 2);
    __bf16* c2s  = (__bf16*)alloc((size_t)Ll * 128 * 128 * 2);
    __bf16* ils  = (__bf16*)alloc((size_t)Ll * 128 * 128 * 2);
    __bf16* l1s  = (__bf16*)alloc(128 * 64 * 2);
    __bf16* l2s  = (__bf16*)alloc(64 * 64 * 2);

    // Weight conversion + B-fragment swizzle (bf16)
    for (int l = 0; l < Ll; l++) {
        { int tot = 2 * 8 * 512;
          k_swizzle<<<(tot + 255) / 256, 256, 0, stream>>>(mlp_w1 + (size_t)l * Gg * Ff,
                                                           w1s + (size_t)l * 64 * 128, Gg, 8, tot); }
        { int tot = 4 * 8 * 512;
          k_swizzle<<<(tot + 255) / 256, 256, 0, stream>>>(mlp_w2 + (size_t)l * 128 * 128,
                                                           w2s + (size_t)l * 16384, 128, 8, tot); }
        { int tot = 4 * 8 * 512;
          k_swizzle<<<(tot + 255) / 256, 256, 0, stream>>>(conv1_w + (size_t)l * 128 * 128,
                                                           c1s + (size_t)l * 16384, 128, 8, tot); }
        { int tot = 4 * 8 * 512;
          k_swizzle<<<(tot + 255) / 256, 256, 0, stream>>>(conv2_w + (size_t)l * 128 * 128,
                                                           c2s + (size_t)l * 16384, 128, 8, tot); }
        { int tot = 4 * 8 * 512;
          k_swizzle<<<(tot + 255) / 256, 256, 0, stream>>>(int_lin_w + (size_t)l * 128 * 128,
                                                           ils + (size_t)l * 16384, 128, 8, tot); }
    }
    { int tot = 4 * 4 * 512;
      k_swizzle<<<(tot + 255) / 256, 256, 0, stream>>>(lin1_w, l1s, 128, 4, tot); }
    { int tot = 2 * 4 * 512;
      k_swizzle<<<(tot + 255) / 256, 256, 0, stream>>>(lin2_w, l2s, 64, 4, tot); }

    // Embedding gather
    { int tot = Nn * Hh;
      k_embed<<<(tot + 255) / 256, 256, 0, stream>>>(x_atoms, embedding, h, tot); }
    k_zero<<<1, 64, 0, stream>>>(hacc, 64);

    const int ntilesN = Nn / 16;   // 3125
    const int ntilesE = Ee / 16;   // 50000
    dim3 nodeGrid((ntilesN + 3) / 4);
    dim3 edgeGrid((ntilesE + 3) / 4);

    for (int l = 0; l < Ll; l++) {
        k_xf<<<nodeGrid, 128, 0, stream>>>(h, c1s + (size_t)l * 16384, xf, ntilesN);
        { int tot = Nn * Ff;
          k_zero<<<(tot + 255) / 256, 256, 0, stream>>>(agg, tot); }
        k_edge<<<edgeGrid, 128, 0, stream>>>(edge_attr, edge_index, xf, agg,
                                             w1s + (size_t)l * 64 * 128,
                                             w2s + (size_t)l * 16384,
                                             mlp_b1 + (size_t)l * Ff,
                                             mlp_b2 + (size_t)l * Ff, ntilesE);
        k_node_update<<<nodeGrid, 128, 0, stream>>>(agg, c2s + (size_t)l * 16384,
                                                    conv2_b + (size_t)l * Hh,
                                                    ils + (size_t)l * 16384,
                                                    int_lin_b + (size_t)l * Hh, h, ntilesN);
    }
    k_head<<<nodeGrid, 128, 0, stream>>>(h, l1s, lin1_b, l2s, lin2_b, hacc, ntilesN);
    k_final<<<1, 32, 0, stream>>>(hacc, readout_w, readout_b, out);
}
